// BahdanauAttention_103079215206
// MI455X (gfx1250) — compile-verified, matrix operand order checked by reading, not run on hardware
//
#include <hip/hip_runtime.h>
#include <hip/hip_bf16.h>

typedef float v2f __attribute__((ext_vector_type(2)));
typedef float v4f __attribute__((ext_vector_type(4)));
typedef float v8f __attribute__((ext_vector_type(8)));
typedef __bf16 bf16_t;
typedef bf16_t v4bf  __attribute__((ext_vector_type(4)));
typedef bf16_t v8bf  __attribute__((ext_vector_type(8)));
typedef bf16_t v16bf __attribute__((ext_vector_type(16)));

// ---------------------------------------------------------------------------
// WMMA wrappers
// ---------------------------------------------------------------------------
__device__ __forceinline__ v8f wmma_f32k4(v2f a, v2f b, v8f c) {
    return __builtin_amdgcn_wmma_f32_16x16x4_f32(
        false, a, false, b, (short)0, c, false, false);
}

__device__ __forceinline__ v8f wmma_bf16(v16bf a, v16bf b, v8f c) {
    return __builtin_amdgcn_wmma_f32_16x16x32_bf16(
        false, a, false, b, (short)0, c, false, false);
}

// 16-bit A/B fragment for 16x16x32: lane half 0 -> K 0..7 & 16..23,
// lane half 1 -> K 8..15 & 24..31 (two contiguous 16B chunks per lane).
__device__ __forceinline__ v16bf load_frag(const bf16_t* base, int k0, int half) {
    v8bf lo = *(const v8bf*)(base + k0 + half * 8);
    v8bf hi = *(const v8bf*)(base + k0 + 16 + half * 8);
    return __builtin_shufflevector(lo, hi, 0, 1, 2, 3, 4, 5, 6, 7,
                                   8, 9, 10, 11, 12, 13, 14, 15);
}

__device__ __forceinline__ float fast_tanh(float x) {
#if __has_builtin(__builtin_amdgcn_tanhf)
    return __builtin_amdgcn_tanhf(x);          // native V_TANH_F32 on gfx1250
#else
    float ax = __builtin_fabsf(x);
    float e  = __builtin_amdgcn_exp2f(ax * -2.88539008177792681f);
    float r  = (1.0f - e) * __builtin_amdgcn_rcpf(1.0f + e);
    return __builtin_copysignf(r, x);
#endif
}

// ---------------------------------------------------------------------------
// fp32 -> bf16 hi/lo split, same (row-major) layout. 1M elements, 4/thread.
// ---------------------------------------------------------------------------
__global__ __launch_bounds__(256) void convert_bf16(
    const float* __restrict__ in, bf16_t* __restrict__ hi,
    bf16_t* __restrict__ lo) {
    const int i0 = (blockIdx.x * 256 + threadIdx.x) * 4;
    v4f x = *(const v4f*)(in + i0);
    v4bf h = __builtin_convertvector(x, v4bf);
    v4f r = x - __builtin_convertvector(h, v4f);
    *(v4bf*)(hi + i0) = h;
    *(v4bf*)(lo + i0) = __builtin_convertvector(r, v4bf);
}

// ---------------------------------------------------------------------------
// fp32 W (1024x1024, row-major [k][n]) -> bf16 hi/lo transposed ([n][k]).
// 32x32 LDS tile transpose, conflict-free (33-float pitch), coalesced R/W.
// ---------------------------------------------------------------------------
__global__ __launch_bounds__(256) void transpose_convert_bf16(
    const float* __restrict__ in, bf16_t* __restrict__ out_hi,
    bf16_t* __restrict__ out_lo) {
    __shared__ float tile[32][33];
    const int tx = threadIdx.x & 31;
    const int ty = threadIdx.x >> 5;            // 0..7
    const int bk = blockIdx.x & 31;             // k-tile
    const int bn = blockIdx.x >> 5;             // n-tile
#pragma unroll
    for (int i = 0; i < 4; ++i)
        tile[ty + 8 * i][tx] = in[(bk * 32 + ty + 8 * i) * 1024 + bn * 32 + tx];
    __syncthreads();
#pragma unroll
    for (int i = 0; i < 4; ++i) {
        float x = tile[tx][ty + 8 * i];
        bf16_t h = (bf16_t)x;
        float r = x - (float)h;
        const int o = (bn * 32 + ty + 8 * i) * 1024 + bk * 32 + tx;
        out_hi[o] = h;
        out_lo[o] = (bf16_t)r;
    }
}

// ---------------------------------------------------------------------------
// P = A @ B via bf16x3 split:  P ~= AhiBhi + AhiBlo + AloBhi  (f32 accum).
// A row-major [m][k], BT column-major [n][k]; M=N=K=1024.
// One wave per 16x16 tile; 32 K-chunks x 3 WMMA(16x16x32_bf16).
// ---------------------------------------------------------------------------
__global__ __launch_bounds__(256) void gemm_bf16x3(
    const bf16_t* __restrict__ Ahi, const bf16_t* __restrict__ Alo,
    const bf16_t* __restrict__ BThi, const bf16_t* __restrict__ BTlo,
    float* __restrict__ P) {
    constexpr int N = 1024, K = 1024;
    const int wid  = threadIdx.x >> 5;
    const int lane = threadIdx.x & 31;
    const int tile = blockIdx.x * 8 + wid;
    const int tM = tile >> 6;
    const int tN = tile & 63;
    const int half = lane >> 4;
    const int mn   = lane & 15;

    const bf16_t* ah = Ahi + (tM * 16 + mn) * K;
    const bf16_t* al = Alo + (tM * 16 + mn) * K;
    const bf16_t* bh = BThi + (tN * 16 + mn) * K;
    const bf16_t* bl = BTlo + (tN * 16 + mn) * K;

    v8f acc = {};
#pragma unroll 2
    for (int k0 = 0; k0 < K; k0 += 32) {
        v16bf fah = load_frag(ah, k0, half);
        v16bf fal = load_frag(al, k0, half);
        v16bf fbh = load_frag(bh, k0, half);
        v16bf fbl = load_frag(bl, k0, half);
        acc = wmma_bf16(fah, fbh, acc);
        acc = wmma_bf16(fah, fbl, acc);
        acc = wmma_bf16(fal, fbh, acc);
    }

    float* out = P + (tM * 16 + half * 8) * N + tN * 16 + mn;
#pragma unroll
    for (int r = 0; r < 8; ++r) out[r * N] = acc[r];
}

// ---------------------------------------------------------------------------
// scores[b,t,s] = sum_u scale[u] * tanh(P1[b,t,u] + P2[b,s,u])
// Block = (b, 16-row t-tile). Each thread owns (t, two s rows) and streams
// u with float4 loads; p1/scale loads amortize over the two s accumulators.
// ---------------------------------------------------------------------------
__global__ __launch_bounds__(256) void scores_kernel(
    const float* __restrict__ P1, const float* __restrict__ P2,
    const float* __restrict__ scale, float* __restrict__ S) {
    const int b  = blockIdx.x >> 3;
    const int tt = blockIdx.x & 7;
    const int t  = tt * 16 + (threadIdx.x >> 4);
    const int sl = threadIdx.x & 15;

    const float* p1 = P1 + (b * 128 + t) * 1024;

    for (int sc = 0; sc < 4; ++sc) {
        const int s0 = sc * 16 + sl;
        const int s1 = s0 + 64;
        const float* p2a = P2 + (b * 128 + s0) * 1024;
        const float* p2b = P2 + (b * 128 + s1) * 1024;
        float acc0 = 0.0f, acc1 = 0.0f;
#pragma unroll 2
        for (int u = 0; u < 1024; u += 4) {
            v4f a  = *(const v4f*)(p1 + u);
            v4f ka = *(const v4f*)(p2a + u);
            v4f kb = *(const v4f*)(p2b + u);
            v4f sv = *(const v4f*)(scale + u);
#pragma unroll
            for (int j = 0; j < 4; ++j) {
                acc0 = fmaf(sv[j], fast_tanh(a[j] + ka[j]), acc0);
                acc1 = fmaf(sv[j], fast_tanh(a[j] + kb[j]), acc1);
            }
        }
        S[(b * 128 + t) * 128 + s0] = acc0;
        S[(b * 128 + t) * 128 + s1] = acc1;
    }
}

// ---------------------------------------------------------------------------
// Masked, numerically-stable softmax over s. One block per (b,t) row.
// ---------------------------------------------------------------------------
__global__ __launch_bounds__(128) void softmax_kernel(
    const float* __restrict__ S, const unsigned char* __restrict__ mask,
    float* __restrict__ attn) {
    __shared__ float red[128];
    const int row = blockIdx.x;        // b*128 + t
    const int b   = row >> 7;
    const int s   = threadIdx.x;

    float v = S[row * 128 + s];
    if (!mask[b * 128 + s]) v = -1e9f;

    red[s] = v;
    __syncthreads();
    for (int st = 64; st > 0; st >>= 1) {
        if (s < st) red[s] = fmaxf(red[s], red[s + st]);
        __syncthreads();
    }
    const float m = red[0];
    __syncthreads();

    const float e = __builtin_amdgcn_exp2f((v - m) * 1.4426950408889634f);
    red[s] = e;
    __syncthreads();
    for (int st = 64; st > 0; st >>= 1) {
        if (s < st) red[s] += red[s + st];
        __syncthreads();
    }
    attn[row * 128 + s] = e / red[0];
}

// ---------------------------------------------------------------------------
// context[b] = attn[b] (128x128) @ value[b] (128x1024), exact fp32 WMMA.
// ---------------------------------------------------------------------------
__global__ __launch_bounds__(256) void context_wmma(
    const float* __restrict__ attn, const float* __restrict__ value,
    float* __restrict__ ctx) {
    const int wid  = threadIdx.x >> 5;
    const int lane = threadIdx.x & 31;
    const int tile = blockIdx.x * 8 + wid;
    const int b  = tile >> 9;
    const int r2 = tile & 511;
    const int tt = r2 >> 6;
    const int vt = r2 & 63;
    const int half = lane >> 4;
    const int mn   = lane & 15;

    const float* aptr = attn  + b * 128 * 128 + (tt * 16 + mn) * 128 + half * 2;
    const float* bptr = value + b * 128 * 1024 + (half * 2) * 1024 + vt * 16 + mn;

    v8f acc = {};
#pragma unroll 4
    for (int k = 0; k < 128; k += 4) {
        v2f a = *(const v2f*)(aptr + k);
        v2f bb;
        bb[0] = bptr[(k + 0) * 1024];
        bb[1] = bptr[(k + 1) * 1024];
        acc = wmma_f32k4(a, bb, acc);
    }

    float* out = ctx + (b * 128 + tt * 16 + half * 8) * 1024 + vt * 16 + mn;
#pragma unroll
    for (int r = 0; r < 8; ++r) out[r * 1024] = acc[r];
}

// ---------------------------------------------------------------------------
extern "C" void kernel_launch(void* const* d_in, const int* in_sizes, int n_in,
                              void* d_out, int out_size, void* d_ws, size_t ws_size,
                              hipStream_t stream) {
    (void)in_sizes; (void)n_in; (void)out_size; (void)ws_size;

    const float*         query = (const float*)d_in[0];         // (8,128,1024)
    const float*         value = (const float*)d_in[1];         // (8,128,1024)
    const unsigned char* mask  = (const unsigned char*)d_in[2]; // (8,128) bool
    const float*         W1    = (const float*)d_in[3];         // (1024,1024)
    const float*         W2    = (const float*)d_in[4];         // (1024,1024)
    const float*         scale = (const float*)d_in[5];         // (1024,)

    char* ws = (char*)d_ws;
    float*  P1    = (float*)(ws);                      // 4 MB
    float*  P2    = (float*)(ws + (size_t)( 4 << 20)); // 4 MB
    float*  S     = (float*)(ws + (size_t)( 8 << 20)); // 0.5 MB
    bf16_t* Qhi   = (bf16_t*)(ws + (size_t)( 9 << 20));
    bf16_t* Qlo   = (bf16_t*)(ws + (size_t)(11 << 20));
    bf16_t* Vhi   = (bf16_t*)(ws + (size_t)(13 << 20));
    bf16_t* Vlo   = (bf16_t*)(ws + (size_t)(15 << 20));
    bf16_t* W1Thi = (bf16_t*)(ws + (size_t)(17 << 20));
    bf16_t* W1Tlo = (bf16_t*)(ws + (size_t)(19 << 20));
    bf16_t* W2Thi = (bf16_t*)(ws + (size_t)(21 << 20));
    bf16_t* W2Tlo = (bf16_t*)(ws + (size_t)(23 << 20));

    float* ctx  = (float*)d_out;                // (8,128,1024)
    float* attn = ctx + 8 * 128 * 1024;         // (8,128,128)

    convert_bf16<<<1024, 256, 0, stream>>>(query, Qhi, Qlo);
    convert_bf16<<<1024, 256, 0, stream>>>(value, Vhi, Vlo);
    transpose_convert_bf16<<<1024, 256, 0, stream>>>(W1, W1Thi, W1Tlo);
    transpose_convert_bf16<<<1024, 256, 0, stream>>>(W2, W2Thi, W2Tlo);

    gemm_bf16x3<<<512, 256, 0, stream>>>(Qhi, Qlo, W1Thi, W1Tlo, P1);
    gemm_bf16x3<<<512, 256, 0, stream>>>(Vhi, Vlo, W2Thi, W2Tlo, P2);

    scores_kernel<<<64, 256, 0, stream>>>(P1, P2, scale, S);
    softmax_kernel<<<1024, 128, 0, stream>>>(S, mask, attn);
    context_wmma<<<512, 256, 0, stream>>>(attn, value, ctx);
}
